// EquivariantAttention_34600256536907
// MI455X (gfx1250) — matrix operand order repaired and checked by reading, not compiled, not run on hardware
//
#include <hip/hip_runtime.h>
#include <hip/hip_bf16.h>
#include <math.h>

typedef __attribute__((ext_vector_type(16))) _Float16 v16h;
typedef __attribute__((ext_vector_type(8)))  float    v8f;
typedef __attribute__((ext_vector_type(4)))  int      v4i;
typedef _Float16 h16;

#define DEV __device__ __forceinline__

constexpr int Bb = 2, Nn = 384, DIMc = 256, Hh = 4, DHd = 64, PD = 128, EDe = 16;
constexpr int BN  = Bb * Nn;     // 768
constexpr int HD  = Hh * DHd;    // 256
constexpr int QKV = 3 * HD;      // 768
constexpr float QSCALE = 0.125f; // DH^-0.5

DEV float sigm(float x) { return 1.0f / (1.0f + __expf(-x)); }
DEV float silu_f(float x) { return x * sigm(x); }
DEV float gelu_f(float x) { return 0.5f * x * (1.0f + erff(x * 0.70710678118654752f)); }

// ------------- async global->LDS staging (gfx1250 path, guarded) -------------
#if defined(__has_builtin)
#if __has_builtin(__builtin_amdgcn_global_load_async_to_lds_b128) && \
    __has_builtin(__builtin_amdgcn_s_wait_asynccnt)
#define USE_ASYNC_COPY 1
#endif
#endif

typedef __attribute__((address_space(1))) v4i g_v4i;   // global int4
typedef __attribute__((address_space(3))) v4i l_v4i;   // LDS int4

// copy `halves` f16 elements (multiple of 8) from global to LDS, block-cooperative
DEV void stage_to_lds(h16* lds_dst, const h16* gsrc, int halves, int tid, int nthreads) {
  int nvec = halves >> 3; // 16B chunks
#ifdef USE_ASYNC_COPY
  for (int v = tid; v < nvec; v += nthreads) {
    __builtin_amdgcn_global_load_async_to_lds_b128(
        (g_v4i*)(gsrc + (size_t)v * 8), (l_v4i*)(lds_dst + (size_t)v * 8), 0, 0);
  }
  __builtin_amdgcn_s_wait_asynccnt(0);
#else
  for (int v = tid; v < nvec; v += nthreads) {
    ((uint4*)lds_dst)[v] = ((const uint4*)gsrc)[v];
  }
#endif
}

// ---------------- WMMA fragment helpers (CDNA5 wave32 layouts) ----------------
// A (16x32 f16): lane<16 -> row=lane, K 0..7 in h0..7, K 16..23 in h8..15
//                lane>=16 -> row=lane-16, K 8..15 and 24..31
DEV v16h load_A16(const h16* base, int lda, int lane) {
  int m = lane & 15;
  int ko = (lane < 16) ? 0 : 8;
  const h16* p = base + m * lda;
  v16h a;
#pragma unroll
  for (int t = 0; t < 8; ++t) a[t] = p[ko + t];
#pragma unroll
  for (int t = 0; t < 8; ++t) a[8 + t] = p[16 + ko + t];
  return a;
}
// B (32x16 f16) from N-major ("BT") storage [N][K]: lane n holds column n;
// lane<16 -> K 0..7 and 16..23, lane>=16 -> K 8..15 and 24..31
DEV v16h load_B16(const h16* baseT, int ldb, int lane) {
  int n = lane & 15;
  int ko = (lane < 16) ? 0 : 8;
  const h16* p = baseT + n * ldb + ko;
  v16h b;
#pragma unroll
  for (int t = 0; t < 8; ++t) b[t] = p[t];
#pragma unroll
  for (int t = 0; t < 8; ++t) b[8 + t] = p[16 + t];
  return b;
}
// C/D (16x16 f32): lane<16 -> col=lane rows 0..7; lane>=16 -> col=lane-16 rows 8..15
DEV v8f wmma_tile(const h16* A, int lda, const h16* BT, int ldb, int K, int lane) {
  v8f c = {};
  for (int k0 = 0; k0 < K; k0 += 32) {
    v16h a = load_A16(A + k0, lda, lane);
    v16h b = load_B16(BT + k0, ldb, lane);
    c = __builtin_amdgcn_wmma_f32_16x16x32_f16(false, a, false, b, (short)0, c, false, false);
  }
  return c;
}

// ---------------- K0: f32 [K][N] -> f16 transposed [N][K] ----------------
__global__ void k_cvtT(const float* __restrict__ src, h16* __restrict__ dst, int K, int N) {
  int idx = blockIdx.x * blockDim.x + threadIdx.x;
  int tot = K * N;
  for (; idx < tot; idx += gridDim.x * blockDim.x) {
    int k = idx / N, n = idx % N;
    dst[n * K + k] = (h16)src[idx];
  }
}

// ---------------- K1: LayerNorm(feats)*gamma + gates ----------------
__global__ void __launch_bounds__(256) k_ln(const float* __restrict__ feats,
                                            const float* __restrict__ gamma,
                                            const float* __restrict__ w_gate,
                                            const float* __restrict__ b_gate,
                                            h16* __restrict__ x16, float* __restrict__ gates) {
  __shared__ float sx[DIMc];
  __shared__ float red[DIMc];
  int r = blockIdx.x, t = threadIdx.x;
  float x = feats[r * DIMc + t];
  red[t] = x; __syncthreads();
  for (int s = DIMc / 2; s > 0; s >>= 1) { if (t < s) red[t] += red[t + s]; __syncthreads(); }
  float mean = red[0] / DIMc; __syncthreads();
  float d = x - mean;
  red[t] = d * d; __syncthreads();
  for (int s = DIMc / 2; s > 0; s >>= 1) { if (t < s) red[t] += red[t + s]; __syncthreads(); }
  float var = red[0] / DIMc;
  float xn = d * rsqrtf(var + 1e-5f) * gamma[t];
  sx[t] = xn;
  x16[r * DIMc + t] = (h16)xn;
  __syncthreads();
  if (t < 2 * Hh) {
    float acc = b_gate[t];
    for (int c = 0; c < DIMc; ++c) acc += sx[c] * w_gate[c * (2 * Hh) + t];
    gates[r * (2 * Hh) + t] = sigm(acc);
  }
}

// ---------------- K2: qkv = x16 @ w_qkv (WMMA), scatter into q/k/vT ----------------
__global__ void __launch_bounds__(256) k_qkv(const h16* __restrict__ x16,
                                             const h16* __restrict__ wqkvT,
                                             h16* __restrict__ q16, h16* __restrict__ k16,
                                             h16* __restrict__ vT16) {
  int lane = threadIdx.x & 31, w = threadIdx.x >> 5;
  int tile = blockIdx.x * 8 + w;        // 48 x 48 tiles
  int rt = tile / 48, ct = tile % 48;
  v8f c = wmma_tile(x16 + rt * 16 * DIMc, DIMc, wqkvT + ct * 16 * DIMc, DIMc, DIMc, lane);
  int col = ct * 16 + (lane & 15);
  int sel = col >> 8, h = (col >> 6) & 3, dd = col & 63;
  int r0 = rt * 16 + ((lane < 16) ? 0 : 8);
#pragma unroll
  for (int r = 0; r < 8; ++r) {
    int row = r0 + r;
    int b = row / Nn, i = row % Nn;
    float v = c[r];
    if (sel == 0)      q16[((b * Hh + h) * Nn + i) * DHd + dd] = (h16)(v * QSCALE);
    else if (sel == 1) k16[((b * Hh + h) * Nn + i) * DHd + dd] = (h16)v;
    else               vT16[((b * Hh + h) * DHd + dd) * Nn + i] = (h16)v;
  }
}

// ---------------- K3: pairwise distances + CoorsNorm ----------------
__global__ void k_pairs(const float* __restrict__ coors, const float* __restrict__ coors_scale,
                        float* __restrict__ rdist, float* __restrict__ rnorm) {
  int idx = blockIdx.x * blockDim.x + threadIdx.x;
  int tot = Bb * Nn * Nn;
  float cs = coors_scale[0];
  for (; idx < tot; idx += gridDim.x * blockDim.x) {
    int b = idx / (Nn * Nn), rem = idx % (Nn * Nn);
    int i = rem / Nn, j = rem % Nn;
    float dx = coors[(b * Nn + i) * 3 + 0] - coors[(b * Nn + j) * 3 + 0];
    float dy = coors[(b * Nn + i) * 3 + 1] - coors[(b * Nn + j) * 3 + 1];
    float dz = coors[(b * Nn + i) * 3 + 2] - coors[(b * Nn + j) * 3 + 2];
    float d = sqrtf(dx * dx + dy * dy + dz * dz);
    rdist[idx] = d;
    float inv = cs / fmaxf(d, 1e-8f);
    rnorm[idx * 3 + 0] = dx * inv;
    rnorm[idx * 3 + 1] = dy * inv;
    rnorm[idx * 3 + 2] = dz * inv;
  }
}

// ---------------- K4: scores = q @ k^T (WMMA) ----------------
__global__ void __launch_bounds__(256) k_qk(const h16* __restrict__ q16,
                                            const h16* __restrict__ k16,
                                            float* __restrict__ scores) {
  int lane = threadIdx.x & 31, w = threadIdx.x >> 5;
  int t = blockIdx.x * 8 + w;              // B*H*24*24 tiles
  int bh = t / (24 * 24); int rem = t % (24 * 24);
  int it = rem / 24, jt = rem % 24;
  v8f c = wmma_tile(q16 + (bh * Nn + it * 16) * DHd, DHd,
                    k16 + (bh * Nn + jt * 16) * DHd, DHd, DHd, lane);
  int jc = jt * 16 + (lane & 15);
  int i0 = it * 16 + ((lane < 16) ? 0 : 8);
#pragma unroll
  for (int r = 0; r < 8; ++r) scores[(bh * Nn + i0 + r) * Nn + jc] = c[r];
}

// ---------------- pos-bias MLP building blocks (LDS staged, one wave per 16 pairs) ----
DEV void ln_silu_rows(float (*act)[PD], h16 (*a16)[PD],
                      const float* __restrict__ g, const float* __restrict__ be, int lane) {
  if (lane < 16) {
    int m = lane;
    float s = 0.f, ss = 0.f;
    for (int c = 0; c < PD; ++c) { float v = act[m][c]; s += v; ss += v * v; }
    float mean = s / PD;
    float var = ss / PD - mean * mean;
    float rstd = rsqrtf(var + 1e-5f);
    for (int c = 0; c < PD; ++c) {
      float v = (act[m][c] - mean) * rstd * g[c] + be[c];
      v = silu_f(v);
      act[m][c] = v;
      a16[m][c] = (h16)v;
    }
  }
}
DEV void mlp_layer(float (*act)[PD], h16 (*a16)[PD], const h16* __restrict__ wT,
                   const float* __restrict__ bias, int lane) {
  // act[16][128] = a16[16][128] @ W[128][128] + bias   (32 WMMAs, B from LDS)
#pragma unroll
  for (int nt = 0; nt < PD / 16; ++nt) {
    v8f c = wmma_tile(&a16[0][0], PD, wT + nt * 16 * PD, PD, PD, lane);
    int col = nt * 16 + (lane & 15);
    int r0 = (lane < 16) ? 0 : 8;
#pragma unroll
    for (int r = 0; r < 8; ++r) act[r0 + r][col] = c[r] + bias[col];
  }
}

// ---------------- K5: pos-bias MLP pass 1 -> scores += qk_pos ----------------
constexpr int PB_TPW = 8;   // tiles per wave
__global__ void __launch_bounds__(128) k_posbias(
    const float* __restrict__ rdist,
    const float* __restrict__ pb_w0, const float* __restrict__ pb_b0,
    const float* __restrict__ pb_g0, const float* __restrict__ pb_be0,
    const h16* __restrict__ pbw1T, const float* __restrict__ pb_b1,
    const float* __restrict__ pb_g1, const float* __restrict__ pb_be1,
    const h16* __restrict__ pbw2T, const float* __restrict__ pb_b2,
    const float* __restrict__ pb_g2, const float* __restrict__ pb_be2,
    const float* __restrict__ w_qkpos, const float* __restrict__ b_qkpos,
    float* __restrict__ scores) {
  extern __shared__ char smem[];
  h16*   w1   = (h16*)smem;                       // 128x128 f16
  h16*   w2   = w1 + PD * PD;                     // 128x128 f16
  float* actb = (float*)(w2 + PD * PD);           // 4 x 16 x 128 f32
  h16*   a16b = (h16*)(actb + 4 * 16 * PD);       // 4 x 16 x 128 f16
  int tid = threadIdx.x;
  int lane = tid & 31, w = tid >> 5;
  float (*act)[PD] = (float (*)[PD])(actb + w * 16 * PD);
  h16   (*a16)[PD] = (h16 (*)[PD])(a16b + w * 16 * PD);

  stage_to_lds(w1, pbw1T, PD * PD, tid, 128);
  stage_to_lds(w2, pbw2T, PD * PD, tid, 128);
  __syncthreads();

  for (int tt = 0; tt < PB_TPW; ++tt) {
    int tile = (blockIdx.x * 4 + w) * PB_TPW + tt;   // B*N*24 tiles total
    int b = tile / (Nn * 24); int rem = tile % (Nn * 24);
    int i = rem / 24; int j0 = (rem % 24) * 16;

    for (int idx = lane; idx < 16 * PD; idx += 32) {
      int m = idx >> 7, c = idx & (PD - 1);
      float d = rdist[(b * Nn + i) * Nn + j0 + m];
      act[m][c] = d * pb_w0[c] + pb_b0[c];
    }
    __syncthreads();
    ln_silu_rows(act, a16, pb_g0, pb_be0, lane);
    __syncthreads();
    mlp_layer(act, a16, w1, pb_b1, lane);
    __syncthreads();
    ln_silu_rows(act, a16, pb_g1, pb_be1, lane);
    __syncthreads();
    mlp_layer(act, a16, w2, pb_b2, lane);
    __syncthreads();
    ln_silu_rows(act, a16, pb_g2, pb_be2, lane);
    __syncthreads();
    if (lane < 16) {
      int m = lane, j = j0 + m;
      for (int hh = 0; hh < Hh; ++hh) {
        float acc = b_qkpos[hh];
        for (int c = 0; c < PD; ++c) acc += act[m][c] * w_qkpos[c * Hh + hh];
        scores[((b * Hh + hh) * Nn + i) * Nn + j] += acc;
      }
    }
    __syncthreads();
  }
}

// ---------------- K6: edge MLP -> cmi (overwrites scores), coor_weights, rel_sign ----
__global__ void k_edge(const float* __restrict__ edges,
                       const float* __restrict__ w_e1, const float* __restrict__ w_e2,
                       const float* __restrict__ w_c, const float* __restrict__ w_cg,
                       const float* __restrict__ b_cg,
                       float* __restrict__ scores, float* __restrict__ cw,
                       float* __restrict__ rs) {
  int idx = blockIdx.x * blockDim.x + threadIdx.x;
  int tot = Bb * Nn * Nn;
  for (; idx < tot; idx += gridDim.x * blockDim.x) {
    int b = idx / (Nn * Nn), rem = idx % (Nn * Nn);
    int i = rem / Nn, j = rem % Nn;
    float t[Hh + EDe];
    for (int h = 0; h < Hh; ++h) t[h] = scores[((b * Hh + h) * Nn + i) * Nn + j];
    for (int e = 0; e < EDe; ++e) t[Hh + e] = edges[(size_t)idx * EDe + e];
    float hid[2 * (Hh + EDe)];
    for (int u = 0; u < 2 * (Hh + EDe); ++u) {
      float acc = 0.f;
      for (int c = 0; c < Hh + EDe; ++c) acc += t[c] * w_e1[c * (2 * (Hh + EDe)) + u];
      hid[u] = gelu_f(acc);
    }
    float cmi[Hh];
    for (int h = 0; h < Hh; ++h) {
      float acc = 0.f;
      for (int u = 0; u < 2 * (Hh + EDe); ++u) acc += hid[u] * w_e2[u * Hh + h];
      cmi[h] = acc;
    }
    for (int h = 0; h < Hh; ++h) scores[((b * Hh + h) * Nn + i) * Nn + j] = cmi[h];
    for (int h = 0; h < Hh; ++h) {
      float a = 0.f, s2 = 0.f;
      for (int g = 0; g < Hh; ++g) {
        a  += gelu_f(cmi[g]) * w_c[g * Hh + h];
        s2 += cmi[g] * w_cg[g * Hh + h];
      }
      cw[(size_t)idx * Hh + h] = a;
      rs[(size_t)idx * Hh + h] = s2 + b_cg[h];
    }
  }
}

// ---------------- K7: softmax over j + talking heads -> attn16 ----------------
__global__ void __launch_bounds__(128) k_softmax(const float* __restrict__ scores,
                                                 const float* __restrict__ w_th,
                                                 h16* __restrict__ attn16) {
  __shared__ float p[Hh][Nn];
  __shared__ float red[128];
  int b = blockIdx.x / Nn, i = blockIdx.x % Nn;
  int t = threadIdx.x;
  for (int g = 0; g < Hh; ++g) {
    const float* row = scores + ((b * Hh + g) * Nn + i) * Nn;
    float m = -3.0e38f;
    for (int j = t; j < Nn; j += 128) m = fmaxf(m, row[j]);
    red[t] = m; __syncthreads();
    for (int s = 64; s > 0; s >>= 1) { if (t < s) red[t] = fmaxf(red[t], red[t + s]); __syncthreads(); }
    m = red[0]; __syncthreads();
    float ssum = 0.f;
    for (int j = t; j < Nn; j += 128) { float e = __expf(row[j] - m); p[g][j] = e; ssum += e; }
    red[t] = ssum; __syncthreads();
    for (int s = 64; s > 0; s >>= 1) { if (t < s) red[t] += red[t + s]; __syncthreads(); }
    float inv = 1.f / red[0]; __syncthreads();
    for (int j = t; j < Nn; j += 128) p[g][j] *= inv;
    __syncthreads();
  }
  float wth[Hh * Hh];
#pragma unroll
  for (int kk = 0; kk < Hh * Hh; ++kk) wth[kk] = w_th[kk];
  for (int j = t; j < Nn; j += 128) {
#pragma unroll
    for (int h = 0; h < Hh; ++h) {
      float acc = 0.f;
#pragma unroll
      for (int g = 0; g < Hh; ++g) acc += wth[h * Hh + g] * p[g][j];
      attn16[((b * Hh + h) * Nn + i) * Nn + j] = (h16)acc;
    }
  }
}

// ---------------- K8: node1 = attn @ v (WMMA) ----------------
__global__ void __launch_bounds__(256) k_attnv(const h16* __restrict__ attn16,
                                               const h16* __restrict__ vT16,
                                               float* __restrict__ node1) {
  int lane = threadIdx.x & 31, w = threadIdx.x >> 5;
  int t = blockIdx.x * 8 + w;            // B*H * 24 * 4 tiles
  int bh = t / (24 * 4); int rem = t % (24 * 4);
  int it = rem / 4, dt = rem % 4;
  v8f c = wmma_tile(attn16 + (size_t)(bh * Nn + it * 16) * Nn, Nn,
                    vT16 + (size_t)(bh * DHd + dt * 16) * Nn, Nn, Nn, lane);
  int dcol = dt * 16 + (lane & 15);
  int i0 = it * 16 + ((lane < 16) ? 0 : 8);
#pragma unroll
  for (int r = 0; r < 8; ++r) node1[(bh * Nn + i0 + r) * DHd + dcol] = c[r];
}

// ---------------- K9: pass 2 — recompute p, s = sum_j attn * p ----------------
__global__ void __launch_bounds__(128) k_pass2(
    const float* __restrict__ rdist, const h16* __restrict__ attn16,
    const float* __restrict__ pb_w0, const float* __restrict__ pb_b0,
    const float* __restrict__ pb_g0, const float* __restrict__ pb_be0,
    const h16* __restrict__ pbw1T, const float* __restrict__ pb_b1,
    const float* __restrict__ pb_g1, const float* __restrict__ pb_be1,
    const h16* __restrict__ pbw2T, const float* __restrict__ pb_b2,
    const float* __restrict__ pb_g2, const float* __restrict__ pb_be2,
    float* __restrict__ sbuf) {
  extern __shared__ char smem[];
  h16*   w1   = (h16*)smem;                         // 128x128 f16
  h16*   w2   = w1 + PD * PD;                       // 128x128 f16
  float* actb = (float*)(w2 + PD * PD);             // 4 x 16 x 128 f32
  h16*   a16b = (h16*)(actb + 4 * 16 * PD);         // 4 x 16 x 128 f16
  float* ssb  = (float*)(a16b + 4 * 16 * PD);       // 4 x 4 x 128 f32
  float* satb = ssb + 4 * Hh * PD;                  // 4 x 4 x 16 f32
  int tid = threadIdx.x;
  int lane = tid & 31, w = tid >> 5;
  float (*act)[PD] = (float (*)[PD])(actb + w * 16 * PD);
  h16   (*a16)[PD] = (h16 (*)[PD])(a16b + w * 16 * PD);
  float (*ss)[PD]  = (float (*)[PD])(ssb + w * Hh * PD);
  float (*sat)[16] = (float (*)[16])(satb + w * Hh * 16);

  stage_to_lds(w1, pbw1T, PD * PD, tid, 128);
  stage_to_lds(w2, pbw2T, PD * PD, tid, 128);

  int row = blockIdx.x * 4 + w;          // one (b,i) row per wave
  int b = row / Nn, i = row % Nn;
  for (int idx = lane; idx < Hh * PD; idx += 32) {
    int hh = idx >> 7, c = idx & (PD - 1);
    ss[hh][c] = 0.f;
  }
  __syncthreads();
  for (int jt = 0; jt < Nn / 16; ++jt) {
    int j0 = jt * 16;
    for (int idx = lane; idx < 16 * PD; idx += 32) {
      int m = idx >> 7, c = idx & (PD - 1);
      float d = rdist[(b * Nn + i) * Nn + j0 + m];
      act[m][c] = d * pb_w0[c] + pb_b0[c];
    }
    for (int idx = lane; idx < Hh * 16; idx += 32) {
      int hh = idx >> 4, m = idx & 15;
      sat[hh][m] = (float)attn16[((b * Hh + hh) * Nn + i) * Nn + j0 + m];
    }
    __syncthreads();
    ln_silu_rows(act, a16, pb_g0, pb_be0, lane);
    __syncthreads();
    mlp_layer(act, a16, w1, pb_b1, lane);
    __syncthreads();
    ln_silu_rows(act, a16, pb_g1, pb_be1, lane);
    __syncthreads();
    mlp_layer(act, a16, w2, pb_b2, lane);
    __syncthreads();
    ln_silu_rows(act, a16, pb_g2, pb_be2, lane);
    __syncthreads();
    for (int c = lane; c < PD; c += 32) {
#pragma unroll
      for (int hh = 0; hh < Hh; ++hh) {
        float acc = ss[hh][c];
#pragma unroll
        for (int m = 0; m < 16; ++m) acc += sat[hh][m] * act[m][c];
        ss[hh][c] = acc;
      }
    }
    __syncthreads();
  }
  for (int idx = lane; idx < Hh * PD; idx += 32) {
    int hh = idx >> 7, c = idx & (PD - 1);
    sbuf[((size_t)(b * Nn + i) * Hh + hh) * PD + c] = ss[hh][c];
  }
}

// ---------------- K10: node2 = s @ w_vpos + rowsum(w_th)*b_vpos; gate; -> no16 ----
__global__ void __launch_bounds__(256) k_node2(const float* __restrict__ node1,
                                               const float* __restrict__ sbuf,
                                               const float* __restrict__ w_vpos,
                                               const float* __restrict__ b_vpos,
                                               const float* __restrict__ w_th,
                                               const float* __restrict__ gates,
                                               h16* __restrict__ no16) {
  int bn = blockIdx.x, t = threadIdx.x;
  int h = t >> 6, dd = t & 63;
  int b = bn / Nn, i = bn % Nn;
  const float* s = sbuf + ((size_t)bn * Hh + h) * PD;
  float acc = 0.f;
  for (int c = 0; c < PD; ++c) acc += s[c] * w_vpos[c * HD + h * DHd + dd];
  float asum = w_th[h * Hh + 0] + w_th[h * Hh + 1] + w_th[h * Hh + 2] + w_th[h * Hh + 3];
  float n1 = node1[((b * Hh + h) * Nn + i) * DHd + dd];
  float v = (n1 + acc + asum * b_vpos[h * DHd + dd]) * gates[bn * (2 * Hh) + h];
  no16[bn * HD + h * DHd + dd] = (h16)v;
}

// ---------------- K11: coordinate branch -> out_coors ----------------
__global__ void __launch_bounds__(128) k_coors(const float* __restrict__ cw,
                                               const float* __restrict__ rs,
                                               const float* __restrict__ rnorm,
                                               const float* __restrict__ gates,
                                               const float* __restrict__ coors_combine,
                                               float* __restrict__ out_coors) {
  __shared__ float red[128];
  __shared__ float mh[Hh], sh[Hh];
  __shared__ float acc[Hh * 3];
  int bn = blockIdx.x, t = threadIdx.x;
  size_t base = (size_t)bn * Nn;
  for (int h = 0; h < Hh; ++h) {
    float m = -3.0e38f;
    for (int j = t; j < Nn; j += 128) m = fmaxf(m, cw[(base + j) * Hh + h]);
    red[t] = m; __syncthreads();
    for (int s = 64; s > 0; s >>= 1) { if (t < s) red[t] = fmaxf(red[t], red[t + s]); __syncthreads(); }
    if (t == 0) mh[h] = red[0];
    __syncthreads();
    float ssum = 0.f;
    for (int j = t; j < Nn; j += 128) ssum += __expf(cw[(base + j) * Hh + h] - mh[h]);
    red[t] = ssum; __syncthreads();
    for (int s = 64; s > 0; s >>= 1) { if (t < s) red[t] += red[t + s]; __syncthreads(); }
    if (t == 0) sh[h] = red[0];
    __syncthreads();
  }
  float la[Hh][3] = {};
  for (int j = t; j < Nn; j += 128) {
    float rn0 = rnorm[(base + j) * 3 + 0];
    float rn1 = rnorm[(base + j) * 3 + 1];
    float rn2 = rnorm[(base + j) * 3 + 2];
#pragma unroll
    for (int h = 0; h < Hh; ++h) {
      float wv = __expf(cw[(base + j) * Hh + h] - mh[h]) / sh[h] * rs[(base + j) * Hh + h];
      la[h][0] += wv * rn0; la[h][1] += wv * rn1; la[h][2] += wv * rn2;
    }
  }
  // deterministic tree reductions for the 12 accumulators
  for (int h = 0; h < Hh; ++h) {
    for (int c = 0; c < 3; ++c) {
      red[t] = la[h][c]; __syncthreads();
      for (int s = 64; s > 0; s >>= 1) { if (t < s) red[t] += red[t + s]; __syncthreads(); }
      if (t == 0) acc[h * 3 + c] = red[0];
      __syncthreads();
    }
  }
  if (t < 3) {
    float o = 0.f;
    for (int h = 0; h < Hh; ++h)
      o += coors_combine[h] * gates[bn * (2 * Hh) + Hh + h] * acc[h * 3 + t];
    out_coors[bn * 3 + t] = o;
  }
}

// ---------------- K12: final projection (WMMA) -> out_nodes ----------------
__global__ void __launch_bounds__(256) k_out(const h16* __restrict__ no16,
                                             const h16* __restrict__ woutT,
                                             const float* __restrict__ b_out,
                                             float* __restrict__ out_nodes) {
  int lane = threadIdx.x & 31, w = threadIdx.x >> 5;
  int t = blockIdx.x * 8 + w;       // 48 x 16 tiles
  int rt = t / 16, ct = t % 16;
  v8f c = wmma_tile(no16 + rt * 16 * DIMc, DIMc, woutT + ct * 16 * DIMc, DIMc, DIMc, lane);
  int col = ct * 16 + (lane & 15);
  int r0 = rt * 16 + ((lane < 16) ? 0 : 8);
#pragma unroll
  for (int r = 0; r < 8; ++r) out_nodes[(r0 + r) * DIMc + col] = c[r] + b_out[col];
}

// ---------------- host launch ----------------
extern "C" void kernel_launch(void* const* d_in, const int* in_sizes, int n_in,
                              void* d_out, int out_size, void* d_ws, size_t ws_size,
                              hipStream_t stream) {
  (void)in_sizes; (void)n_in; (void)out_size; (void)ws_size;
  const float* feats         = (const float*)d_in[0];
  const float* coors         = (const float*)d_in[1];
  const float* edges         = (const float*)d_in[2];
  const float* gamma         = (const float*)d_in[3];
  const float* w_qkv         = (const float*)d_in[4];
  const float* w_gate        = (const float*)d_in[5];
  const float* b_gate        = (const float*)d_in[6];
  const float* w_th          = (const float*)d_in[7];
  const float* w_e1          = (const float*)d_in[8];
  const float* w_e2          = (const float*)d_in[9];
  const float* w_c           = (const float*)d_in[10];
  const float* w_cg          = (const float*)d_in[11];
  const float* b_cg          = (const float*)d_in[12];
  const float* coors_scale   = (const float*)d_in[13];
  const float* coors_combine = (const float*)d_in[14];
  const float* pb_w0  = (const float*)d_in[15];
  const float* pb_b0  = (const float*)d_in[16];
  const float* pb_g0  = (const float*)d_in[17];
  const float* pb_be0 = (const float*)d_in[18];
  const float* pb_w1  = (const float*)d_in[19];
  const float* pb_b1  = (const float*)d_in[20];
  const float* pb_g1  = (const float*)d_in[21];
  const float* pb_be1 = (const float*)d_in[22];
  const float* pb_w2  = (const float*)d_in[23];
  const float* pb_b2  = (const float*)d_in[24];
  const float* pb_g2  = (const float*)d_in[25];
  const float* pb_be2 = (const float*)d_in[26];
  const float* w_qkpos = (const float*)d_in[27];
  const float* b_qkpos = (const float*)d_in[28];
  const float* w_vpos  = (const float*)d_in[29];
  const float* b_vpos  = (const float*)d_in[30];
  const float* w_out   = (const float*)d_in[31];
  const float* b_out   = (const float*)d_in[32];

  char* wsp = (char*)d_ws;
  size_t off = 0;
  auto alloc = [&](size_t bytes) -> void* {
    void* p = wsp + off;
    off += (bytes + 255) & ~(size_t)255;
    return p;
  };
  h16* x16    = (h16*)alloc((size_t)BN * DIMc * 2);
  h16* wqkvT  = (h16*)alloc((size_t)QKV * DIMc * 2);
  h16* pbw1T  = (h16*)alloc((size_t)PD * PD * 2);
  h16* pbw2T  = (h16*)alloc((size_t)PD * PD * 2);
  h16* woutT  = (h16*)alloc((size_t)DIMc * DIMc * 2);
  h16* q16    = (h16*)alloc((size_t)Bb * Hh * Nn * DHd * 2);
  h16* k16    = (h16*)alloc((size_t)Bb * Hh * Nn * DHd * 2);
  h16* vT16   = (h16*)alloc((size_t)Bb * Hh * DHd * Nn * 2);
  h16* attn16 = (h16*)alloc((size_t)Bb * Hh * Nn * Nn * 2);
  h16* no16   = (h16*)alloc((size_t)BN * HD * 2);
  float* gates  = (float*)alloc((size_t)BN * 2 * Hh * 4);
  float* scores = (float*)alloc((size_t)Bb * Hh * Nn * Nn * 4);
  float* rdist  = (float*)alloc((size_t)Bb * Nn * Nn * 4);
  float* rnorm  = (float*)alloc((size_t)Bb * Nn * Nn * 3 * 4);
  float* cw     = (float*)alloc((size_t)Bb * Nn * Nn * Hh * 4);
  float* rs     = (float*)alloc((size_t)Bb * Nn * Nn * Hh * 4);
  float* sbuf   = (float*)alloc((size_t)BN * Hh * PD * 4);
  float* node1  = (float*)alloc((size_t)Bb * Hh * Nn * DHd * 4);

  float* out_nodes = (float*)d_out;
  float* out_coors = (float*)d_out + (size_t)BN * DIMc;

  // weight conversion / transposition to f16
  k_cvtT<<<(DIMc * QKV + 255) / 256, 256, 0, stream>>>(w_qkv, wqkvT, DIMc, QKV);
  k_cvtT<<<(PD * PD + 255) / 256, 256, 0, stream>>>(pb_w1, pbw1T, PD, PD);
  k_cvtT<<<(PD * PD + 255) / 256, 256, 0, stream>>>(pb_w2, pbw2T, PD, PD);
  k_cvtT<<<(DIMc * DIMc + 255) / 256, 256, 0, stream>>>(w_out, woutT, DIMc, DIMc);

  k_ln<<<BN, 256, 0, stream>>>(feats, gamma, w_gate, b_gate, x16, gates);
  k_qkv<<<(48 * 48) / 8, 256, 0, stream>>>(x16, wqkvT, q16, k16, vT16);
  k_pairs<<<1152, 256, 0, stream>>>(coors, coors_scale, rdist, rnorm);
  k_qk<<<(Bb * Hh * 24 * 24) / 8, 256, 0, stream>>>(q16, k16, scores);

  size_t sh_pos = (size_t)2 * PD * PD * 2            // w1 + w2
                + (size_t)4 * 16 * PD * 4            // act
                + (size_t)4 * 16 * PD * 2;           // a16
  k_posbias<<<(Bb * Nn * 24) / (4 * PB_TPW), 128, sh_pos, stream>>>(
      rdist, pb_w0, pb_b0, pb_g0, pb_be0, pbw1T, pb_b1, pb_g1, pb_be1,
      pbw2T, pb_b2, pb_g2, pb_be2, w_qkpos, b_qkpos, scores);

  k_edge<<<1152, 256, 0, stream>>>(edges, w_e1, w_e2, w_c, w_cg, b_cg, scores, cw, rs);
  k_softmax<<<BN, 128, 0, stream>>>(scores, w_th, attn16);
  k_attnv<<<(Bb * Hh * 24 * 4) / 8, 256, 0, stream>>>(attn16, vT16, node1);

  size_t sh_p2 = (size_t)2 * PD * PD * 2             // w1 + w2
               + (size_t)4 * 16 * PD * 4             // act
               + (size_t)4 * 16 * PD * 2             // a16
               + (size_t)4 * Hh * PD * 4             // ss
               + (size_t)4 * Hh * 16 * 4;            // sat
  k_pass2<<<BN / 4, 128, sh_p2, stream>>>(
      rdist, attn16, pb_w0, pb_b0, pb_g0, pb_be0, pbw1T, pb_b1, pb_g1, pb_be1,
      pbw2T, pb_b2, pb_g2, pb_be2, sbuf);

  k_node2<<<BN, 256, 0, stream>>>(node1, sbuf, w_vpos, b_vpos, w_th, gates, no16);
  k_coors<<<BN, 128, 0, stream>>>(cw, rs, rnorm, gates, coors_combine, out_coors);
  k_out<<<(48 * 16) / 8, 256, 0, stream>>>(no16, woutT, b_out, out_nodes);
}
